// MultiheadDiffAttn_21268678050071
// MI455X (gfx1250) — matrix operand order repaired and checked by reading, not compiled, not run on hardware
//
#include <hip/hip_runtime.h>
#include <hip/hip_bf16.h>
#include <stdint.h>

// ---------------------------------------------------------------------------
// Types for CDNA5 WMMA (wave32)
// ---------------------------------------------------------------------------
typedef __bf16 bf16_t;
typedef __attribute__((ext_vector_type(16))) __bf16 v16bf;
typedef __attribute__((ext_vector_type(8)))  float  f32x8;

#define T_SEQ   2048
#define E_DIM   2048
#define H_HEADS 16
#define D_HEAD  64
#define LAMBDA_INIT_C 0.783605766531624f   // 0.8 - 0.6*exp(-0.3*12)
#define SCALING_C     0.125f               // 64^-0.5
#define EPS_C         1e-5f

static __device__ __forceinline__ f32x8 wmma_bf16(v16bf a, v16bf b, f32x8 c) {
    // D = A(16x32 bf16) x B(32x16 bf16) + C(16x16 f32)
    return __builtin_amdgcn_wmma_f32_16x16x32_bf16(false, a, false, b,
                                                   (short)0, c, false, false);
}

// gfx1250 async LDS DMA: per-lane 16B global -> LDS, tracked by ASYNCcnt.
// LDS aperture: low 32 bits of a generic pointer to __shared__ are the LDS
// byte offset, which is what VDST must hold.
static __device__ __forceinline__ void async_load_b128(void* lds_ptr, const void* gptr) {
    uint32_t lds_addr = (uint32_t)(uintptr_t)lds_ptr;
    uint64_t gaddr    = (uint64_t)(uintptr_t)gptr;
    asm volatile("global_load_async_to_lds_b128 %0, %1, off"
                 :: "v"(lds_addr), "v"(gaddr) : "memory");
}
static __device__ __forceinline__ void wait_async0() {
    asm volatile("s_wait_asynccnt 0" ::: "memory");
}

// A-fragment (16x32 bf16). p0 points at row 0 of the 16-row tile; `stride` in
// elements. Per ISA: lane L (m=L&15, h=L>>4) holds row m, K = 8h..8h+7 in
// elems 0..7 and K = 16+8h..16+8h+7 in elems 8..15.
static __device__ __forceinline__ v16bf frag_a(const bf16_t* p0, int stride, int lane) {
    const int m = lane & 15, h = (lane >> 4) & 1;
    const bf16_t* p = p0 + (size_t)m * stride;
    union { v16bf v; uint4 u[2]; } r;
    r.u[0] = *reinterpret_cast<const uint4*>(p + 8 * h);
    r.u[1] = *reinterpret_cast<const uint4*>(p + 16 + 8 * h);
    return r.v;
}

// B-fragment (32x16 bf16), source stored column-major-by-row: row n of the
// source array is column n of B, contraction index contiguous.
// Lane L (c=L&15, h=L>>4) holds column c, K = 8h..8h+7 in elems 0..7 and K = 16+8h..16+8h+7 in elems 8..15.
static __device__ __forceinline__ v16bf frag_b(const bf16_t* p0, int stride, int lane) {
    const int c = lane & 15, h = (lane >> 4) & 1;
    const bf16_t* p = p0 + (size_t)c * stride;
    union { v16bf v; uint4 u[2]; } r;
    r.u[0] = *reinterpret_cast<const uint4*>(p + 8 * h);
    r.u[1] = *reinterpret_cast<const uint4*>(p + 16 + 8 * h);
    return r.v;
}

static __device__ __forceinline__ f32x8 zero8() {
    f32x8 z;
#pragma unroll
    for (int i = 0; i < 8; ++i) z[i] = 0.0f;
    return z;
}

// ---------------------------------------------------------------------------
// fp32 -> bf16 conversion (vectorized x4)
// ---------------------------------------------------------------------------
__global__ void cvt_f32_bf16_kernel(const float* __restrict__ s,
                                    bf16_t* __restrict__ d, int n4) {
    int i = blockIdx.x * blockDim.x + threadIdx.x;
    if (i >= n4) return;
    float4 f = reinterpret_cast<const float4*>(s)[i];
    union { uint2 u; bf16_t e[4]; } o;
    o.e[0] = (bf16_t)f.x; o.e[1] = (bf16_t)f.y;
    o.e[2] = (bf16_t)f.z; o.e[3] = (bf16_t)f.w;
    reinterpret_cast<uint2*>(d)[i] = o.u;
}

// ---------------------------------------------------------------------------
// lambda_full = exp(lq1.lk1) - exp(lq2.lk2) + lambda_init
// ---------------------------------------------------------------------------
__global__ void lambda_kernel(const float* __restrict__ lq1, const float* __restrict__ lk1,
                              const float* __restrict__ lq2, const float* __restrict__ lk2,
                              float* __restrict__ lam) {
    __shared__ float s1[64], s2[64];
    int t = threadIdx.x;
    s1[t] = lq1[t] * lk1[t];
    s2[t] = lq2[t] * lk2[t];
    __syncthreads();
    if (t == 0) {
        float a = 0.f, b = 0.f;
        for (int i = 0; i < 64; ++i) { a += s1[i]; b += s2[i]; }
        lam[0] = __expf(a) - __expf(b) + LAMBDA_INIT_C;
    }
}

// ---------------------------------------------------------------------------
// bf16 WMMA GEMM: C[M,N] = alpha * A[M,K] @ B[N,K]^T   (both row-major, K contig)
// block = 128 threads (4 waves), block tile 128x128, wave tile 64x64,
// K-step 32, double-buffered LDS filled by async LDS-DMA (no VGPR staging).
// ---------------------------------------------------------------------------
template <bool OUT_F32>
__global__ __launch_bounds__(128)
void gemm_bf16_wmma(const bf16_t* __restrict__ A, const bf16_t* __restrict__ B,
                    void* __restrict__ Cout, int M, int N, int K, float alpha) {
    __shared__ alignas(16) bf16_t As[2 * 128 * 40];
    __shared__ alignas(16) bf16_t Bs[2 * 128 * 40];

    const int tid  = threadIdx.x;
    const int lane = tid & 31;
    const int w    = tid >> 5;           // 0..3
    const int wr   = (w & 1) * 64;       // wave-tile row offset
    const int wc   = (w >> 1) * 64;      // wave-tile col offset
    const int m0   = blockIdx.y * 128;
    const int n0   = blockIdx.x * 128;

    f32x8 acc[4][4];
#pragma unroll
    for (int r = 0; r < 4; ++r)
#pragma unroll
        for (int c = 0; c < 4; ++c) acc[r][c] = zero8();

    // staging: 128x32 tile = 256 x 16B; thread covers rows sr0, sr0+64 at col sc0
    const int sr0 = tid >> 1;
    const int sc0 = (tid & 1) * 16;

    const bf16_t* Abase = A + (size_t)m0 * K;
    const bf16_t* Bbase = B + (size_t)n0 * K;

    auto stage = [&](int k0, int b) {   // async DMA tile into LDS buffer b
        bf16_t* as = As + b * (128 * 40);
        bf16_t* bs = Bs + b * (128 * 40);
        async_load_b128(as + sr0 * 40 + sc0,        Abase + (size_t)sr0 * K + k0 + sc0);
        async_load_b128(as + (sr0 + 64) * 40 + sc0, Abase + (size_t)(sr0 + 64) * K + k0 + sc0);
        async_load_b128(bs + sr0 * 40 + sc0,        Bbase + (size_t)sr0 * K + k0 + sc0);
        async_load_b128(bs + (sr0 + 64) * 40 + sc0, Bbase + (size_t)(sr0 + 64) * K + k0 + sc0);
    };

    stage(0, 0);
    wait_async0();
    __syncthreads();

    const int niter = K >> 5;
    int buf = 0;
    for (int it = 0; it < niter; ++it) {
        if (it + 1 < niter) stage((it + 1) << 5, buf ^ 1);  // DMA next tile
        if (it + 2 < niter) {                               // gfx1250 global_prefetch
            const int k2 = (it + 2) << 5;
            __builtin_prefetch(Abase + (size_t)sr0 * K + k2 + sc0, 0, 1);
            __builtin_prefetch(Bbase + (size_t)sr0 * K + k2 + sc0, 0, 1);
        }

        const bf16_t* as = As + buf * (128 * 40);
        const bf16_t* bs = Bs + buf * (128 * 40);
        v16bf af[4], bfr[4];
#pragma unroll
        for (int i = 0; i < 4; ++i) af[i]  = frag_a(as + (wr + i * 16) * 40, 40, lane);
#pragma unroll
        for (int i = 0; i < 4; ++i) bfr[i] = frag_b(bs + (wc + i * 16) * 40, 40, lane);
#pragma unroll
        for (int rt = 0; rt < 4; ++rt)
#pragma unroll
            for (int ct = 0; ct < 4; ++ct)
                acc[rt][ct] = wmma_bf16(af[rt], bfr[ct], acc[rt][ct]);

        if (it + 1 < niter) wait_async0();   // next tile resident in LDS
        __syncthreads();
        buf ^= 1;
    }

    // C layout: elem j of v8f = row j + 8*(lane>>4); col = lane&15
    const int rofs = 8 * (lane >> 4);
    const int cofs = lane & 15;
#pragma unroll
    for (int rt = 0; rt < 4; ++rt)
#pragma unroll
        for (int ct = 0; ct < 4; ++ct)
#pragma unroll
            for (int j = 0; j < 8; ++j) {
                int row = m0 + wr + rt * 16 + j + rofs;
                int col = n0 + wc + ct * 16 + cofs;
                float v = alpha * acc[rt][ct][j];
                if (OUT_F32)
                    reinterpret_cast<float*>(Cout)[(size_t)row * N + col] = v;
                else
                    reinterpret_cast<bf16_t*>(Cout)[(size_t)row * N + col] = (bf16_t)v;
            }
}

// ---------------------------------------------------------------------------
// Differential flash attention.
// grid = (T/64, H). block = 128 threads (4 waves), wave owns 16 q rows, both
// softmax paths (sub-heads 2h, 2h+1 share K sub-head h, V head h/2).
// Q layout [T][2048] bf16 (pre-scaled), K [T][1024], V [T][1024].
// Output: RMS-normed, g-scaled, (1-lambda_init)-scaled bf16 [T][2048].
// ---------------------------------------------------------------------------
__global__ __launch_bounds__(128, 1)
void diff_attn_kernel(const bf16_t* __restrict__ Qb, const bf16_t* __restrict__ Kb,
                      const bf16_t* __restrict__ Vb, const float* __restrict__ g,
                      const float* __restrict__ lam_ptr, bf16_t* __restrict__ Ab) {
    __shared__ alignas(16) bf16_t VsT[128 * 40];      // V^T tile: [vdim 0..127][kv 0..31]
    __shared__ alignas(16) bf16_t Pl[4 * 16 * 32];    // per-wave P tile 16x32

    const int tid  = threadIdx.x;
    const int w    = tid >> 5;
    const int lane = tid & 31;
    const int lm   = lane & 15;
    const int lh   = lane >> 4;
    const int head = blockIdx.y;
    const int qblk = blockIdx.x;
    const int qr0  = qblk * 64 + w * 16;   // wave's first q row
    const int wrow_max = qr0 + 15;

    // Q fragments for both paths (scale already folded into Q)
    const bf16_t* q1base = Qb + (size_t)qr0 * E_DIM + (size_t)(2 * head) * D_HEAD;
    v16bf qf[2][2];
#pragma unroll
    for (int d0 = 0; d0 < 2; ++d0) {
        qf[0][d0] = frag_a(q1base + d0 * 32, E_DIM, lane);
        qf[1][d0] = frag_a(q1base + 64 + d0 * 32, E_DIM, lane);
    }

    f32x8 acc[2][8];
#pragma unroll
    for (int p = 0; p < 2; ++p)
#pragma unroll
        for (int ct = 0; ct < 8; ++ct) acc[p][ct] = zero8();

    float mrow[2][8], lsum[2][8];
#pragma unroll
    for (int p = 0; p < 2; ++p)
#pragma unroll
        for (int j = 0; j < 8; ++j) { mrow[p][j] = -3.0e38f; lsum[p][j] = 0.0f; }

    const int vcb = (head >> 1) * 128;     // V column base within [T][1024]
    const int nch = qblk * 2 + 2;          // causal: kv chunks of 32

    bf16_t* myP = Pl + w * (16 * 32);

    for (int ch = 0; ch < nch; ++ch) {
        const int kv0 = ch * 32;
        __syncthreads();
        // Stage V^T tile: 32 kv rows x 128 dims -> VsT[dim][kv], all threads
#pragma unroll
        for (int it = 0; it < 4; ++it) {
            int u  = tid + it * 128;          // 0..511
            int kv = u >> 4;                  // 0..31
            int c8 = (u & 15) * 8;            // 0..120
            union { uint4 q; bf16_t e[8]; } vv;
            vv.q = *reinterpret_cast<const uint4*>(
                Vb + (size_t)(kv0 + kv) * 1024 + vcb + c8);
#pragma unroll
            for (int e = 0; e < 8; ++e) VsT[(c8 + e) * 40 + kv] = vv.e[e];
        }
        __syncthreads();

        if (kv0 > wrow_max) continue;         // fully masked for this wave

        // K fragments (same K for both paths)
        v16bf kf[2][2];
#pragma unroll
        for (int sc = 0; sc < 2; ++sc)
#pragma unroll
            for (int d0 = 0; d0 < 2; ++d0)
                kf[sc][d0] = frag_b(Kb + (size_t)(kv0 + sc * 16) * 1024
                                       + (size_t)head * D_HEAD + d0 * 32, 1024, lane);

#pragma unroll
        for (int p = 0; p < 2; ++p) {
            f32x8 S[2];
#pragma unroll
            for (int sc = 0; sc < 2; ++sc) {
                S[sc] = wmma_bf16(qf[p][0], kf[sc][0], zero8());
                S[sc] = wmma_bf16(qf[p][1], kf[sc][1], S[sc]);
            }
            // online softmax per row (row j+8*lh lives in 16 lanes of a half)
#pragma unroll
            for (int j = 0; j < 8; ++j) {
                const int row = qr0 + j + 8 * lh;
                float s0 = S[0][j], s1 = S[1][j];
                if (kv0 + lm > row)      s0 = -3.0e38f;
                if (kv0 + 16 + lm > row) s1 = -3.0e38f;
                float rx = fmaxf(s0, s1);
                rx = fmaxf(rx, __shfl_xor(rx, 1));
                rx = fmaxf(rx, __shfl_xor(rx, 2));
                rx = fmaxf(rx, __shfl_xor(rx, 4));
                rx = fmaxf(rx, __shfl_xor(rx, 8));
                float mnew = fmaxf(mrow[p][j], rx);
                float corr = __expf(mrow[p][j] - mnew);
                float e0 = __expf(s0 - mnew);
                float e1 = __expf(s1 - mnew);
                float rs = e0 + e1;
                rs += __shfl_xor(rs, 1);
                rs += __shfl_xor(rs, 2);
                rs += __shfl_xor(rs, 4);
                rs += __shfl_xor(rs, 8);
                lsum[p][j] = lsum[p][j] * corr + rs;
                mrow[p][j] = mnew;
#pragma unroll
                for (int ct = 0; ct < 8; ++ct) acc[p][ct][j] *= corr;
                myP[(j + 8 * lh) * 32 + lm]      = (bf16_t)e0;
                myP[(j + 8 * lh) * 32 + 16 + lm] = (bf16_t)e1;
            }
            asm volatile("s_wait_dscnt 0" ::: "memory");  // wave-private LDS RAW
            v16bf pf = frag_a(myP, 32, lane);
            // V fragments re-read from LDS per use (limits live VGPRs)
#pragma unroll
            for (int ct = 0; ct < 8; ++ct) {
                v16bf vfr = frag_b(VsT + (ct * 16) * 40, 40, lane);
                acc[p][ct] = wmma_bf16(pf, vfr, acc[p][ct]);
            }
        }
    }

    // Epilogue: combine paths, RMSNorm over 128, scale by g and (1-lambda_init)
    const float lam    = lam_ptr[0];
    const float oscale = 1.0f - LAMBDA_INIT_C;
    float gl[8];
#pragma unroll
    for (int ct = 0; ct < 8; ++ct) gl[ct] = g[ct * 16 + lm];

#pragma unroll
    for (int j = 0; j < 8; ++j) {
        const float i1 = 1.0f / lsum[0][j];
        const float i2 = lam / lsum[1][j];
        float yv[8];
        float ss = 0.0f;
#pragma unroll
        for (int ct = 0; ct < 8; ++ct) {
            float v = acc[0][ct][j] * i1 - acc[1][ct][j] * i2;
            yv[ct] = v;
            ss += v * v;
        }
        ss += __shfl_xor(ss, 1);
        ss += __shfl_xor(ss, 2);
        ss += __shfl_xor(ss, 4);
        ss += __shfl_xor(ss, 8);
        const float r = rsqrtf(ss * (1.0f / 128.0f) + EPS_C);
        const int row = qr0 + j + 8 * lh;
        bf16_t* op = Ab + (size_t)row * E_DIM + (size_t)head * 128;
#pragma unroll
        for (int ct = 0; ct < 8; ++ct)
            op[ct * 16 + lm] = (bf16_t)(yv[ct] * r * gl[ct] * oscale);
    }
}

// ---------------------------------------------------------------------------
// Host launcher
// ---------------------------------------------------------------------------
extern "C" void kernel_launch(void* const* d_in, const int* in_sizes, int n_in,
                              void* d_out, int out_size, void* d_ws, size_t ws_size,
                              hipStream_t stream) {
    (void)in_sizes; (void)n_in; (void)out_size; (void)ws_size;
    const float* x   = (const float*)d_in[0];
    const float* Wq  = (const float*)d_in[2];
    const float* Wk  = (const float*)d_in[3];
    const float* Wv  = (const float*)d_in[4];
    const float* Wo  = (const float*)d_in[5];
    const float* lq1 = (const float*)d_in[6];
    const float* lk1 = (const float*)d_in[7];
    const float* lq2 = (const float*)d_in[8];
    const float* lk2 = (const float*)d_in[9];
    const float* g   = (const float*)d_in[10];

    char* ws = (char*)d_ws;
    const size_t MB = 1024ull * 1024ull;
    bf16_t* Xb  = (bf16_t*)(ws + 0 * MB);     // 8 MB  x bf16
    bf16_t* Wqb = (bf16_t*)(ws + 8 * MB);     // 8 MB
    bf16_t* Wkb = (bf16_t*)(ws + 16 * MB);    // 4 MB
    bf16_t* Wvb = (bf16_t*)(ws + 20 * MB);    // 4 MB
    bf16_t* Wob = (bf16_t*)(ws + 24 * MB);    // 8 MB
    bf16_t* Qb  = (bf16_t*)(ws + 32 * MB);    // 8 MB  [T][2048], pre-scaled
    bf16_t* Kb  = (bf16_t*)(ws + 40 * MB);    // 4 MB  [T][1024]
    bf16_t* Vb  = (bf16_t*)(ws + 44 * MB);    // 4 MB  [T][1024]
    bf16_t* Ab  = (bf16_t*)(ws + 48 * MB);    // 8 MB  normed attn out bf16
    float*  lam = (float*)(ws + 56 * MB);     // 4 B

    auto cvt = [&](const float* s, bf16_t* d, int n) {
        int n4 = n / 4;
        cvt_f32_bf16_kernel<<<(n4 + 255) / 256, 256, 0, stream>>>(s, d, n4);
    };
    cvt(x,  Xb,  T_SEQ * E_DIM);
    cvt(Wq, Wqb, E_DIM * E_DIM);
    cvt(Wk, Wkb, (E_DIM / 2) * E_DIM);
    cvt(Wv, Wvb, (E_DIM / 2) * E_DIM);
    cvt(Wo, Wob, E_DIM * E_DIM);

    lambda_kernel<<<1, 64, 0, stream>>>(lq1, lk1, lq2, lk2, lam);

    // Q = (x @ Wq^T) * scaling ; K = x @ Wk^T ; V = x @ Wv^T
    gemm_bf16_wmma<false><<<dim3(16, 16), 128, 0, stream>>>(
        Xb, Wqb, (void*)Qb, T_SEQ, 2048, E_DIM, SCALING_C);
    gemm_bf16_wmma<false><<<dim3(8, 16), 128, 0, stream>>>(
        Xb, Wkb, (void*)Kb, T_SEQ, 1024, E_DIM, 1.0f);
    gemm_bf16_wmma<false><<<dim3(8, 16), 128, 0, stream>>>(
        Xb, Wvb, (void*)Vb, T_SEQ, 1024, E_DIM, 1.0f);

    // differential flash attention + RMSNorm epilogue
    diff_attn_kernel<<<dim3(T_SEQ / 64, H_HEADS), 128, 0, stream>>>(
        Qb, Kb, Vb, g, lam, Ab);

    // out = attn @ Wo^T  (fp32 output)
    gemm_bf16_wmma<true><<<dim3(16, 16), 128, 0, stream>>>(
        Ab, Wob, d_out, T_SEQ, 2048, E_DIM, 1.0f);
}
